// MemoryAugmentedAttention_54503134986453
// MI455X (gfx1250) — compile-verified, matrix operand order checked by reading, not run on hardware
//
#include <hip/hip_runtime.h>

#define S_LEN 2048
#define E_DIM 1024
#define H_NUM 16
#define D_DIM 64

typedef __attribute__((ext_vector_type(16))) __bf16 v16bf;
typedef __attribute__((ext_vector_type(8)))  float  v8f;
typedef __attribute__((ext_vector_type(4)))  int    v4i_;

#define AS1_ __attribute__((address_space(1)))
#define AS3_ __attribute__((address_space(3)))

#if __has_builtin(__builtin_amdgcn_global_load_async_to_lds_b128)
#define HAVE_ASYNC_LDS 1
#else
#define HAVE_ASYNC_LDS 0
#endif

__device__ __forceinline__ unsigned short f2bf_bits(float f) {
  unsigned u = __builtin_bit_cast(unsigned, f);
  unsigned r = u + 0x7FFFu + ((u >> 16) & 1u);   // round-to-nearest-even
  return (unsigned short)(r >> 16);
}
__device__ __forceinline__ __bf16 bf_from_bits(unsigned short s) {
  return __builtin_bit_cast(__bf16, s);
}
__device__ __forceinline__ float f_from_bits(unsigned short s) {
  unsigned u = ((unsigned)s) << 16;
  return __builtin_bit_cast(float, u);
}

// 16-byte global->LDS copy: async (ASYNCcnt, no VGPR data) when available.
__device__ __forceinline__ void g2lds16(const void* g, void* l) {
#if HAVE_ASYNC_LDS
  __builtin_amdgcn_global_load_async_to_lds_b128(
      (AS1_ v4i_*)g, (AS3_ v4i_*)l, 0, 0);
#else
  *(uint4*)l = *(const uint4*)g;
#endif
}
__device__ __forceinline__ void g2lds_wait() {
#if HAVE_ASYNC_LDS
#if __has_builtin(__builtin_amdgcn_s_wait_asynccnt)
  __builtin_amdgcn_s_wait_asynccnt(0);
#else
  asm volatile("s_wait_asynccnt 0x0" ::: "memory");
#endif
#endif
}

// ---- WMMA fragment gathers (ISA 7.12.2, 16-bit layouts, wave32) ----
// A 16x32: lane row = lane&15; VGPR v holds K = 2*(v&3) + 16*(v>>2) + 8*(lane>>4)
__device__ __forceinline__ v16bf frag_a16(const unsigned short* p, int ld) {
  const int lane = threadIdx.x & 31;
  const int row = lane & 15, hh = (lane >> 4) << 3;
  v16bf a;
#pragma unroll
  for (int v = 0; v < 8; ++v) {
    int kb = ((v & 3) << 1) + ((v >> 2) << 4) + hh;
    a[2 * v]     = bf_from_bits(p[row * ld + kb]);
    a[2 * v + 1] = bf_from_bits(p[row * ld + kb + 1]);
  }
  return a;
}
__device__ __forceinline__ v16bf frag_a16_scaled(const unsigned short* p, int ld, float sc) {
  const int lane = threadIdx.x & 31;
  const int row = lane & 15, hh = (lane >> 4) << 3;
  v16bf a;
#pragma unroll
  for (int v = 0; v < 8; ++v) {
    int kb = ((v & 3) << 1) + ((v >> 2) << 4) + hh;
    a[2 * v]     = bf_from_bits(f2bf_bits(f_from_bits(p[row * ld + kb]) * sc));
    a[2 * v + 1] = bf_from_bits(f2bf_bits(f_from_bits(p[row * ld + kb + 1]) * sc));
  }
  return a;
}
// B 32x16 from row-major [K][N] storage
__device__ __forceinline__ v16bf frag_b16(const unsigned short* p, int ld) {
  const int lane = threadIdx.x & 31;
  const int col = lane & 15, hh = (lane >> 4) << 3;
  v16bf b;
#pragma unroll
  for (int v = 0; v < 8; ++v) {
    int kb = ((v & 3) << 1) + ((v >> 2) << 4) + hh;
    b[2 * v]     = bf_from_bits(p[kb * ld + col]);
    b[2 * v + 1] = bf_from_bits(p[(kb + 1) * ld + col]);
  }
  return b;
}
// B 32x16 where storage is [N][K] (rows are keys): on-the-fly transpose
__device__ __forceinline__ v16bf frag_bT16(const unsigned short* p, int ld) {
  const int lane = threadIdx.x & 31;
  const int col = lane & 15, hh = (lane >> 4) << 3;
  v16bf b;
#pragma unroll
  for (int v = 0; v < 8; ++v) {
    int kb = ((v & 3) << 1) + ((v >> 2) << 4) + hh;
    b[2 * v]     = bf_from_bits(p[col * ld + kb]);
    b[2 * v + 1] = bf_from_bits(p[col * ld + kb + 1]);
  }
  return b;
}
__device__ __forceinline__ v8f wmma_bf16(v16bf a, v16bf b, v8f c) {
  return __builtin_amdgcn_wmma_f32_16x16x32_bf16(false, a, false, b, (short)0, c,
                                                 false, false);
}

// ---- C = A(Mx1024) @ W(1024x1024) + bias ; A fp32 or bf16, C fp32 or bf16 ----
// grid = (M/128)*16 blocks, 256 threads (8 waves). Block tile 128x64.
template <int ABF16, int CF32>
__global__ __launch_bounds__(256) void gemm_bias_1024(
    const void* __restrict__ Aptr, const float* __restrict__ W,
    const float* __restrict__ bias, void* __restrict__ Cptr) {
  constexpr int N = 1024, K = 1024;
  constexpr int LDA = 40;   // multiple of 8 ushorts -> 16B-aligned async rows
  __shared__ __align__(16) unsigned short lA[128 * LDA];
  __shared__ __align__(16) unsigned short lB[32 * 68];
  const int tid = threadIdx.x, wave = tid >> 5, lane = tid & 31;
  const int bm = blockIdx.x >> 4, bn = blockIdx.x & 15;
  const int row0 = bm << 7, n0 = bn << 6;

  v8f acc[4];
#pragma unroll
  for (int t = 0; t < 4; ++t)
#pragma unroll
    for (int i = 0; i < 8; ++i) acc[t][i] = 0.0f;

  for (int k0 = 0; k0 < K; k0 += 32) {
    __syncthreads();
    if (ABF16) {
      const unsigned short* A = (const unsigned short*)Aptr;
#pragma unroll
      for (int i = 0; i < 2; ++i) {
        int idx = tid + (i << 8);                 // 512 x 16B chunks (128x32 bf16)
        int r = idx >> 2, c = (idx & 3) << 3;
        g2lds16(&A[(size_t)(row0 + r) * K + k0 + c], &lA[r * LDA + c]);
      }
    } else {
      const float* A = (const float*)Aptr;
#pragma unroll
      for (int i = 0; i < 4; ++i) {
        int idx = tid + (i << 8);                 // 1024 float4
        int r = idx >> 3, c = (idx & 7) << 2;
        float4 f = *(const float4*)&A[(size_t)(row0 + r) * K + k0 + c];
        unsigned short* d = &lA[r * LDA + c];
        d[0] = f2bf_bits(f.x); d[1] = f2bf_bits(f.y);
        d[2] = f2bf_bits(f.z); d[3] = f2bf_bits(f.w);
      }
    }
#pragma unroll
    for (int i = 0; i < 2; ++i) {
      int idx = tid + (i << 8);                   // 512 float4
      int r = idx >> 4, c = (idx & 15) << 2;
      float4 f = *(const float4*)&W[(size_t)(k0 + r) * N + n0 + c];
      unsigned short* d = &lB[r * 68 + c];
      d[0] = f2bf_bits(f.x); d[1] = f2bf_bits(f.y);
      d[2] = f2bf_bits(f.z); d[3] = f2bf_bits(f.w);
    }
    if (k0 + 32 < K)
      __builtin_prefetch(&W[(size_t)(k0 + 32 + (tid >> 4)) * N + n0 + ((tid & 15) << 2)], 0, 1);
    if (ABF16) g2lds_wait();
    __syncthreads();
    v16bf a = frag_a16(&lA[(wave << 4) * LDA], LDA);
#pragma unroll
    for (int t = 0; t < 4; ++t) {
      v16bf bfr = frag_b16(&lB[t << 4], 68);
      acc[t] = wmma_bf16(a, bfr, acc[t]);
    }
  }

  const int colb = lane & 15, half = lane >> 4;
#pragma unroll
  for (int t = 0; t < 4; ++t) {
    int n = n0 + (t << 4) + colb;
    float bv = bias[n];
#pragma unroll
    for (int r = 0; r < 8; ++r) {
      int row = row0 + (wave << 4) + r + (half << 3);
      float v = acc[t][r] + bv;
      if (CF32) ((float*)Cptr)[(size_t)row * N + n] = v;
      else ((unsigned short*)Cptr)[(size_t)row * N + n] = f2bf_bits(v);
    }
  }
}

// ---- fused flash attention with memory-key bias ----
// grid = B*H*(S/128) = 1024 blocks, 256 threads; wave = one 16-row q tile.
__global__ __launch_bounds__(256) void flash_attn_mem(
    const unsigned short* __restrict__ Q, const unsigned short* __restrict__ Kk,
    const unsigned short* __restrict__ V, const float* __restrict__ mem,
    const int* __restrict__ mask, unsigned short* __restrict__ O) {
  __shared__ __align__(16) unsigned short lQ[128 * 72];
  __shared__ __align__(16) unsigned short lK[32 * 72];
  __shared__ __align__(16) unsigned short lV[32 * 72];
  __shared__ __align__(16) unsigned short lM[32 * 72];
  __shared__ __align__(16) unsigned short sP[8][16 * 34];

  const int tid = threadIdx.x, wave = tid >> 5, lane = tid & 31;
  const int colb = lane & 15, half = lane >> 4;
  const int qb = blockIdx.x & 15, bh = blockIdx.x >> 4;
  const int b = bh >> 4, h = bh & 15;
  const int qblk = qb << 7, q0 = qblk + (wave << 4);

  // stage 128x64 bf16 Q slab (async: 1024 x 16B chunks)
#pragma unroll
  for (int i = 0; i < 4; ++i) {
    int idx = tid + (i << 8);
    int r = idx >> 3, c = (idx & 7) << 3;
    g2lds16(&Q[(size_t)(b * S_LEN + qblk + r) * E_DIM + h * D_DIM + c],
            &lQ[r * 72 + c]);
  }
  g2lds_wait();
  __syncthreads();
  v16bf fq[2], fqs[2];
#pragma unroll
  for (int s = 0; s < 2; ++s) {
    fq[s]  = frag_a16(&lQ[(wave << 4) * 72 + (s << 5)], 72);
    fqs[s] = frag_a16_scaled(&lQ[(wave << 4) * 72 + (s << 5)], 72, 0.125f); // 1/sqrt(64)
  }

  float mrow[8], lrow[8];
  v8f o[4];
#pragma unroll
  for (int r = 0; r < 8; ++r) { mrow[r] = -3.0e38f; lrow[r] = 0.0f; }
#pragma unroll
  for (int n = 0; n < 4; ++n)
#pragma unroll
    for (int r = 0; r < 8; ++r) o[n][r] = 0.0f;

  for (int c0 = 0; c0 < S_LEN; c0 += 32) {
    __syncthreads();
    {   // stage K and V chunks (32x64 bf16 each): one async B128 per thread each
      int r = tid >> 3, c = (tid & 7) << 3;
      g2lds16(&Kk[(size_t)(b * S_LEN + c0 + r) * E_DIM + h * D_DIM + c],
              &lK[r * 72 + c]);
      g2lds16(&V[(size_t)(b * S_LEN + c0 + r) * E_DIM + h * D_DIM + c],
              &lV[r * 72 + c]);
    }
#pragma unroll
    for (int i = 0; i < 2; ++i) {
      int idx = tid + (i << 8);                   // 512 float4 (memory rows, f32->bf16)
      int r = idx >> 4, c = (idx & 15) << 2;
      float4 f = *(const float4*)&mem[(size_t)(c0 + r) * D_DIM + c];
      unsigned short* d = &lM[r * 72 + c];
      d[0] = f2bf_bits(f.x); d[1] = f2bf_bits(f.y);
      d[2] = f2bf_bits(f.z); d[3] = f2bf_bits(f.w);
    }
    g2lds_wait();
    __syncthreads();

    // scores tile 16x32: (Q/8)*K^T + Q*mem^T
    v8f acc[2];
#pragma unroll
    for (int t = 0; t < 2; ++t)
#pragma unroll
      for (int i = 0; i < 8; ++i) acc[t][i] = 0.0f;
#pragma unroll
    for (int t = 0; t < 2; ++t)
#pragma unroll
      for (int s = 0; s < 2; ++s) {
        v16bf kb = frag_bT16(&lK[(t << 4) * 72 + (s << 5)], 72);
        acc[t] = wmma_bf16(fqs[s], kb, acc[t]);
        v16bf mb = frag_bT16(&lM[(t << 4) * 72 + (s << 5)], 72);
        acc[t] = wmma_bf16(fq[s], mb, acc[t]);
      }

    // mask + online softmax (row stats reduced over 16-lane halves)
    float cmax[8];
#pragma unroll
    for (int r = 0; r < 8; ++r) cmax[r] = -3.0e38f;
#pragma unroll
    for (int t = 0; t < 2; ++t)
#pragma unroll
      for (int r = 0; r < 8; ++r) {
        int q = q0 + r + (half << 3);
        int kk = c0 + (t << 4) + colb;
        float sc = acc[t][r];
        if (mask[(size_t)b * S_LEN * S_LEN + (size_t)q * S_LEN + kk] == 0)
          sc = -3.0e38f;
        acc[t][r] = sc;
        cmax[r] = fmaxf(cmax[r], sc);
      }
#pragma unroll
    for (int m = 1; m <= 8; m <<= 1)
#pragma unroll
      for (int r = 0; r < 8; ++r)
        cmax[r] = fmaxf(cmax[r], __shfl_xor(cmax[r], m, 32));

    float alpha[8], rsum[8];
#pragma unroll
    for (int r = 0; r < 8; ++r) {
      float mn = fmaxf(mrow[r], cmax[r]);
      alpha[r] = __expf(mrow[r] - mn);
      mrow[r] = mn;
      rsum[r] = 0.0f;
    }
#pragma unroll
    for (int t = 0; t < 2; ++t)
#pragma unroll
      for (int r = 0; r < 8; ++r) {
        float p = __expf(acc[t][r] - mrow[r]);
        rsum[r] += p;
        sP[wave][(r + (half << 3)) * 34 + (t << 4) + colb] = f2bf_bits(p);
      }
#pragma unroll
    for (int m = 1; m <= 8; m <<= 1)
#pragma unroll
      for (int r = 0; r < 8; ++r) rsum[r] += __shfl_xor(rsum[r], m, 32);
#pragma unroll
    for (int r = 0; r < 8; ++r) lrow[r] = lrow[r] * alpha[r] + rsum[r];
#pragma unroll
    for (int n = 0; n < 4; ++n)
#pragma unroll
      for (int r = 0; r < 8; ++r) o[n][r] *= alpha[r];

    __builtin_amdgcn_wave_barrier();   // order sP stores vs fragment gather
    v16bf fp = frag_a16(&sP[wave][0], 34);
#pragma unroll
    for (int n = 0; n < 4; ++n) {
      v16bf vb = frag_b16(&lV[n << 4], 72);
      o[n] = wmma_bf16(fp, vb, o[n]);
    }
  }

  float inv[8];
#pragma unroll
  for (int r = 0; r < 8; ++r) inv[r] = 1.0f / lrow[r];
#pragma unroll
  for (int n = 0; n < 4; ++n)
#pragma unroll
    for (int r = 0; r < 8; ++r) {
      int q = q0 + r + (half << 3);
      O[(size_t)(b * S_LEN + q) * E_DIM + h * D_DIM + (n << 4) + colb] =
          f2bf_bits(o[n][r] * inv[r]);
    }
}

extern "C" void kernel_launch(void* const* d_in, const int* in_sizes, int n_in,
                              void* d_out, int out_size, void* d_ws, size_t ws_size,
                              hipStream_t stream) {
  (void)in_sizes; (void)n_in; (void)out_size; (void)ws_size;
  const float* query = (const float*)d_in[0];
  const float* key   = (const float*)d_in[1];
  const float* value = (const float*)d_in[2];
  const int*   mask  = (const int*)d_in[3];
  const float* Wq = (const float*)d_in[4];  const float* bq = (const float*)d_in[5];
  const float* Wk = (const float*)d_in[6];  const float* bk = (const float*)d_in[7];
  const float* Wv = (const float*)d_in[8];  const float* bv = (const float*)d_in[9];
  const float* Wo = (const float*)d_in[10]; const float* bo = (const float*)d_in[11];
  const float* memory = (const float*)d_in[12];
  float* out = (float*)d_out;

  const size_t MAT = (size_t)4 * S_LEN * E_DIM;   // 8192*1024 elements
  unsigned short* qws = (unsigned short*)d_ws;
  unsigned short* kws = qws + MAT;
  unsigned short* vws = kws + MAT;
  unsigned short* ows = vws + MAT;

  dim3 blk(256);
  dim3 gGemm(1024);   // (8192/128) * (1024/64)
  gemm_bias_1024<0, 0><<<gGemm, blk, 0, stream>>>(query, Wq, bq, qws);
  gemm_bias_1024<0, 0><<<gGemm, blk, 0, stream>>>(key,   Wk, bk, kws);
  gemm_bias_1024<0, 0><<<gGemm, blk, 0, stream>>>(value, Wv, bv, vws);

  dim3 gAttn(1024);   // B*H * (S/128)
  flash_attn_mem<<<gAttn, blk, 0, stream>>>(qws, kws, vws, memory, mask, ows);

  gemm_bias_1024<1, 1><<<gGemm, blk, 0, stream>>>(ows, Wo, bo, out);
}
